// Seq2SeqRNN_39367670235486
// MI455X (gfx1250) — matrix operand order repaired and checked by reading, not compiled, not run on hardware
//
#include <hip/hip_runtime.h>
#include <stdint.h>

typedef unsigned short ushort_t;
typedef __attribute__((ext_vector_type(16))) __bf16        v16bf;
typedef __attribute__((ext_vector_type(8)))  float         v8f;
typedef __attribute__((ext_vector_type(4)))  unsigned int  v4u;

union FragBf {
    v16bf bf;
    v4u   q[2];   // q[0] = elements 0..7, q[1] = elements 8..15
};

#define Hc    512     // hidden size
#define KS    544     // augmented K: 512 Whh cols + [512]=Wih (input) + [513]=bias + pad
#define HPAD  552     // padded LDS row stride (bf16 elems) -> bank-conflict dodge
#define BT    32      // batch rows per workgroup
#define SEQ   128     // encoder sequence length
#define TOUT  24      // decoder output steps
#define NTHREADS 256  // 8 wave32 per workgroup

// ---------- bf16 helpers ----------
__device__ __forceinline__ ushort_t f2bf(float f) {        // round-to-nearest-even
    unsigned int u = __float_as_uint(f);
    unsigned int r = u + 0x7FFFu + ((u >> 16) & 1u);
    return (ushort_t)(r >> 16);
}
__device__ __forceinline__ float bf2f(ushort_t h) {
    return __uint_as_float(((unsigned int)h) << 16);
}
// CDNA5 hardware tanh (TRANS32 op); trailing v_nop covers the trans-op
// result-use hazard that the compiler cannot see through inline asm.
__device__ __forceinline__ float hw_tanh(float x) {
    float r;
    asm volatile("v_tanh_f32 %0, %1\n\tv_nop" : "=v"(r) : "v"(x));
    return r;
}

// ---------- prologue: build augmented bf16 hi/lo weight matrices ----------
// Layout: W[n*KS + k];  k<512: Whh[n][k];  k==512: Wih[n];  k==513: bih[n]+bhh[n];  else 0
__global__ void __launch_bounds__(256)
build_weights(const float* __restrict__ encW, const float* __restrict__ encWih,
              const float* __restrict__ encBih, const float* __restrict__ encBhh,
              const float* __restrict__ decW, const float* __restrict__ decWih,
              const float* __restrict__ decBih, const float* __restrict__ decBhh,
              ushort_t* __restrict__ ehi, ushort_t* __restrict__ elo,
              ushort_t* __restrict__ dhi, ushort_t* __restrict__ dlo) {
    int i = blockIdx.x * blockDim.x + threadIdx.x;
    if (i >= Hc * KS) return;
    int n = i / KS, k = i % KS;
    float we, wd;
    if (k < Hc)          { we = encW[n * Hc + k];             wd = decW[n * Hc + k]; }
    else if (k == Hc)    { we = encWih[n];                    wd = decWih[n]; }
    else if (k == Hc + 1){ we = encBih[n] + encBhh[n];        wd = decBih[n] + decBhh[n]; }
    else                 { we = 0.f;                          wd = 0.f; }
    ushort_t h = f2bf(we);
    ehi[i] = h;
    elo[i] = f2bf(we - bf2f(h));
    h = f2bf(wd);
    dhi[i] = h;
    dlo[i] = f2bf(wd - bf2f(h));
}

// ---------- fragment load: A (h, LDS) + B (W, global/L2) for one 32-wide K slice ----------
__device__ __forceinline__ void load_frags(
    FragBf ahi[2], FragBf alo[2], FragBf bhi[4], FragBf blo[4],
    const ushort_t* a0, const ushort_t* a1,
    const ushort_t* b0, const ushort_t* b1, int kk)
{
    #pragma unroll
    for (int mt = 0; mt < 2; ++mt) {
        const ushort_t* pa = a0 + mt * (16 * HPAD) + kk;
        const ushort_t* pl = a1 + mt * (16 * HPAD) + kk;
        ahi[mt].q[0] = *(const v4u*)(pa);
        ahi[mt].q[1] = *(const v4u*)(pa + 8);
        alo[mt].q[0] = *(const v4u*)(pl);
        alo[mt].q[1] = *(const v4u*)(pl + 8);
    }
    #pragma unroll
    for (int nt = 0; nt < 4; ++nt) {
        const ushort_t* pw = b0 + (size_t)nt * (16 * KS) + kk;
        const ushort_t* pl = b1 + (size_t)nt * (16 * KS) + kk;
        bhi[nt].q[0] = *(const v4u*)(pw);
        bhi[nt].q[1] = *(const v4u*)(pw + 8);
        blo[nt].q[0] = *(const v4u*)(pl);
        blo[nt].q[1] = *(const v4u*)(pl + 8);
    }
}

// ---------- compensated bf16 product: acc += al*bh + ah*bl + ah*bh ----------
__device__ __forceinline__ void wmma_all(
    v8f acc[2][4], const FragBf ahi[2], const FragBf alo[2],
    const FragBf bhi[4], const FragBf blo[4])
{
    #pragma unroll
    for (int mt = 0; mt < 2; ++mt) {
        #pragma unroll
        for (int nt = 0; nt < 4; ++nt) {
            acc[mt][nt] = __builtin_amdgcn_wmma_f32_16x16x32_bf16(
                false, alo[mt].bf, false, bhi[nt].bf, (short)0, acc[mt][nt], false, false);
            acc[mt][nt] = __builtin_amdgcn_wmma_f32_16x16x32_bf16(
                false, ahi[mt].bf, false, blo[nt].bf, (short)0, acc[mt][nt], false, false);
            acc[mt][nt] = __builtin_amdgcn_wmma_f32_16x16x32_bf16(
                false, ahi[mt].bf, false, bhi[nt].bf, (short)0, acc[mt][nt], false, false);
        }
    }
}

// First slice: SRC2 is a literal zero vector so the compiler folds it to the
// inline 0 operand -> no per-step accumulator-init mov wall.
__device__ __forceinline__ void wmma_first(
    v8f acc[2][4], const FragBf ahi[2], const FragBf alo[2],
    const FragBf bhi[4], const FragBf blo[4])
{
    const v8f z = {};
    #pragma unroll
    for (int mt = 0; mt < 2; ++mt) {
        #pragma unroll
        for (int nt = 0; nt < 4; ++nt) {
            v8f d = __builtin_amdgcn_wmma_f32_16x16x32_bf16(
                false, alo[mt].bf, false, bhi[nt].bf, (short)0, z, false, false);
            d = __builtin_amdgcn_wmma_f32_16x16x32_bf16(
                false, ahi[mt].bf, false, blo[nt].bf, (short)0, d, false, false);
            acc[mt][nt] = __builtin_amdgcn_wmma_f32_16x16x32_bf16(
                false, ahi[mt].bf, false, bhi[nt].bf, (short)0, d, false, false);
        }
    }
}

// ---------- one recurrent step: h_next = tanh(h_aug @ W_aug^T) ----------
// 2-stage software pipeline over 17 K-slices: peel(s0,s1) + 7 double-trips + tail(s16).
__device__ __forceinline__ void rnn_step(
    const ushort_t* hc_hi, const ushort_t* hc_lo,   // LDS, current h  [BT][HPAD]
    ushort_t* hn_hi, ushort_t* hn_lo,               // LDS, next h
    const ushort_t* __restrict__ Whi,               // global, [n*KS + k]
    const ushort_t* __restrict__ Wlo)
{
    const int tid  = threadIdx.x;
    const int lane = tid & 31;
    const int wave = tid >> 5;
    const int l15  = lane & 15;
    const int g    = lane >> 4;     // half-wave select
    const int wnb  = wave * 64;     // this wave's N base (4 tiles of 16)

    // per-lane fragment base pointers (A: row = lane%16, K-half by g;
    //                                  B: col = lane%16, K = e + 16g)
    const ushort_t* a0 = hc_hi + l15 * HPAD + 16 * g;
    const ushort_t* a1 = hc_lo + l15 * HPAD + 16 * g;
    const ushort_t* b0 = Whi + (size_t)(wnb + l15) * KS + 16 * g;
    const ushort_t* b1 = Wlo + (size_t)(wnb + l15) * KS + 16 * g;

    v8f acc[2][4];
    FragBf ah0[2], al0[2], bh0[4], bl0[4];
    FragBf ah1[2], al1[2], bh1[4], bl1[4];

    // peeled prologue: slices 0,1 (slice 0 WMMA folds the zero accumulator)
    load_frags(ah0, al0, bh0, bl0, a0, a1, b0, b1, 0);
    load_frags(ah1, al1, bh1, bl1, a0, a1, b0, b1, 32);
    wmma_first(acc, ah0, al0, bh0, bl0);
    load_frags(ah0, al0, bh0, bl0, a0, a1, b0, b1, 64);
    wmma_all(acc, ah1, al1, bh1, bl1);

    #pragma unroll 1
    for (int j = 1; j < 8; ++j) {
        const int kk = j * 64;
        load_frags(ah1, al1, bh1, bl1, a0, a1, b0, b1, kk + 32);
        wmma_all(acc, ah0, al0, bh0, bl0);
        load_frags(ah0, al0, bh0, bl0, a0, a1, b0, b1, kk + 64);
        wmma_all(acc, ah1, al1, bh1, bl1);
    }
    wmma_all(acc, ah0, al0, bh0, bl0);      // tail: slice 16 (kk = 512..543)

    // tanh (hardware TRANS op) + cheap hi/lo split back into LDS (n < 512 only)
    #pragma unroll
    for (int mt = 0; mt < 2; ++mt) {
        #pragma unroll
        for (int v = 0; v < 8; ++v) {
            int m = mt * 16 + v + 8 * g;            // C layout: M = vgpr + 8*(lane/16)
            #pragma unroll
            for (int nt = 0; nt < 4; ++nt) {
                int n = wnb + nt * 16 + l15;        // C layout: N = lane%16
                float h = hw_tanh(acc[mt][nt][v]);
                unsigned int u = __float_as_uint(h);
                ushort_t hi = (ushort_t)(u >> 16);                          // truncation split
                float hf = __uint_as_float(u & 0xFFFF0000u);
                hn_hi[m * HPAD + n] = hi;
                hn_lo[m * HPAD + n] = f2bf(h - hf);
            }
        }
    }
}

// ---------- LDS footprint ----------
#define HBUF_BYTES (BT * HPAD * 2)                 // one bf16 h buffer: 35328 B
#define SMEM_BYTES (4 * HBUF_BYTES                 /* h hi/lo double-buffered */ \
                    + BT * SEQ * 4                 /* x slab, transposed [t][m] */ \
                    + Hc * 4                       /* fcw */ \
                    + BT * 8 * 4)                  /* fc partials */

__global__ void __launch_bounds__(NTHREADS) __attribute__((amdgpu_waves_per_eu(2)))
seq2seq_kernel(const float* __restrict__ x,
               const float* __restrict__ fc_W, const float* __restrict__ fc_b,
               const ushort_t* __restrict__ encW_hi, const ushort_t* __restrict__ encW_lo,
               const ushort_t* __restrict__ decW_hi, const ushort_t* __restrict__ decW_lo,
               float* __restrict__ out)
{
    extern __shared__ char smem[];
    char* p = smem;
    ushort_t* hhi0 = (ushort_t*)p; p += HBUF_BYTES;
    ushort_t* hlo0 = (ushort_t*)p; p += HBUF_BYTES;
    ushort_t* hhi1 = (ushort_t*)p; p += HBUF_BYTES;
    ushort_t* hlo1 = (ushort_t*)p; p += HBUF_BYTES;
    float* x_lds   = (float*)p;    p += BT * SEQ * 4;   // [t][m]
    float* fcw     = (float*)p;    p += Hc * 4;
    float* part    = (float*)p;    p += BT * 8 * 4;

    const int tid    = threadIdx.x;
    const int wgBase = blockIdx.x * BT;

    // ---- stage: fcW, x (transposed), zero both h buffers ----
    for (int i = tid; i < Hc; i += NTHREADS)
        fcw[i] = fc_W[i];
    for (int i = tid; i < BT * SEQ; i += NTHREADS) {
        int m = i / SEQ, t = i % SEQ;                       // coalesced global read
        x_lds[t * BT + m] = x[(size_t)(wgBase + m) * SEQ + t];
    }
    for (int i = tid; i < BT * HPAD; i += NTHREADS) {
        hhi0[i] = 0; hlo0[i] = 0;
        hhi1[i] = 0; hlo1[i] = 0;
    }
    __syncthreads();

    // augmented columns: [512] = input (step 0: x[m][0]), [513] = 1.0 (both buffers)
    if (tid < BT) {
        const ushort_t one_hi = f2bf(1.0f);
        hhi0[tid * HPAD + (Hc + 1)] = one_hi;
        hhi1[tid * HPAD + (Hc + 1)] = one_hi;
        float inp = x_lds[0 * BT + tid];
        ushort_t hi = f2bf(inp);
        hhi0[tid * HPAD + Hc] = hi;
        hlo0[tid * HPAD + Hc] = f2bf(inp - bf2f(hi));
    }
    __syncthreads();

    ushort_t* hc_hi = hhi0; ushort_t* hc_lo = hlo0;
    ushort_t* hn_hi = hhi1; ushort_t* hn_lo = hlo1;

    // ---- encoder: 128 steps, one barrier per step ----
    #pragma unroll 1
    for (int t = 0; t < SEQ; ++t) {
        rnn_step(hc_hi, hc_lo, hn_hi, hn_lo, encW_hi, encW_lo);
        // input for step t+1 (t==127 -> x[:, -1] = decoder's first input)
        if (tid < BT) {
            int tn = (t + 1 < SEQ) ? (t + 1) : (SEQ - 1);
            float inp = x_lds[tn * BT + tid];
            ushort_t hi = f2bf(inp);
            hn_hi[tid * HPAD + Hc] = hi;
            hn_lo[tid * HPAD + Hc] = f2bf(inp - bf2f(hi));
        }
        { ushort_t* s;
          s = hc_hi; hc_hi = hn_hi; hn_hi = s;
          s = hc_lo; hc_lo = hn_lo; hn_lo = s; }
        __syncthreads();
    }

    // ---- decoder: 24 steps; fc output feeds next step via aug column ----
    #pragma unroll 1
    for (int t = 0; t < TOUT; ++t) {
        rnn_step(hc_hi, hc_lo, hn_hi, hn_lo, decW_hi, decW_lo);
        { ushort_t* s;
          s = hc_hi; hc_hi = hn_hi; hn_hi = s;
          s = hc_lo; hc_lo = hn_lo; hn_lo = s; }
        __syncthreads();

        // fc: out[m] = dot(h2[m,:], fcW) + fcb  (8 threads per row, 64 cols each)
        {
            int m = tid >> 3, c = tid & 7;
            const ushort_t* hh = hc_hi + m * HPAD + c * 64;
            const ushort_t* hl = hc_lo + m * HPAD + c * 64;
            const float*    fw = fcw + c * 64;
            float psum = 0.f;
            for (int j = 0; j < 64; ++j)
                psum += (bf2f(hh[j]) + bf2f(hl[j])) * fw[j];
            part[m * 8 + c] = psum;
        }
        __syncthreads();
        if (tid < BT) {
            float o = fc_b[0];
            #pragma unroll
            for (int c2 = 0; c2 < 8; ++c2) o += part[tid * 8 + c2];
            out[(size_t)(wgBase + tid) * TOUT + t] = o;
            // feedback: next decoder input goes into the buffer read next step
            ushort_t hi = f2bf(o);
            hc_hi[tid * HPAD + Hc] = hi;
            hc_lo[tid * HPAD + Hc] = f2bf(o - bf2f(hi));
        }
        __syncthreads();
    }
}

extern "C" void kernel_launch(void* const* d_in, const int* in_sizes, int n_in,
                              void* d_out, int out_size, void* d_ws, size_t ws_size,
                              hipStream_t stream) {
    (void)in_sizes; (void)n_in; (void)out_size; (void)ws_size;
    const float* x       = (const float*)d_in[0];
    const float* enc_Wih = (const float*)d_in[1];
    const float* enc_Whh = (const float*)d_in[2];
    const float* enc_bih = (const float*)d_in[3];
    const float* enc_bhh = (const float*)d_in[4];
    const float* dec_Wih = (const float*)d_in[5];
    const float* dec_Whh = (const float*)d_in[6];
    const float* dec_bih = (const float*)d_in[7];
    const float* dec_bhh = (const float*)d_in[8];
    const float* fc_W    = (const float*)d_in[9];
    const float* fc_b    = (const float*)d_in[10];

    ushort_t* ehi = (ushort_t*)d_ws;
    ushort_t* elo = ehi + (size_t)Hc * KS;
    ushort_t* dhi = elo + (size_t)Hc * KS;
    ushort_t* dlo = dhi + (size_t)Hc * KS;

    build_weights<<<(Hc * KS + 255) / 256, 256, 0, stream>>>(
        enc_Whh, enc_Wih, enc_bih, enc_bhh,
        dec_Whh, dec_Wih, dec_bih, dec_bhh,
        ehi, elo, dhi, dlo);

    seq2seq_kernel<<<2048 / BT, NTHREADS, SMEM_BYTES, stream>>>(
        x, fc_W, fc_b, ehi, elo, dhi, dlo, (float*)d_out);
}
